// BitLinearMetal_59803124630310
// MI455X (gfx1250) — 8-bit matrix operand order repaired and checked by reading, compiled, not run on hardware
//
#include <hip/hip_runtime.h>
#include <hip/hip_bf16.h>

constexpr int M = 8192;
constexpr int K = 4096;
constexpr int N = 11008;
constexpr int KC = K / 64;
constexpr int NB = N / 16;
constexpr int NTILES_B = NB * KC;

typedef int v8i __attribute__((ext_vector_type(8)));

__device__ __forceinline__ unsigned short f2bf(float f) {
    union { float f; unsigned int u; } c; c.f = f;
    unsigned int r = c.u + 0x7FFFu + ((c.u >> 16) & 1u);
    return (unsigned short)(r >> 16);
}

__global__ __launch_bounds__(256) void quant_pack_A(const float* __restrict__ x,
                                                    unsigned char* __restrict__ qA,
                                                    float* __restrict__ inv_s) {
    const int m = blockIdx.x;
    const int t = threadIdx.x;
    const float* row = x + (size_t)m * K;

    float xv[16];
    float mx = 0.0f;
    const float4* p = (const float4*)(row + t * 16);
#pragma unroll
    for (int i = 0; i < 4; ++i) {
        float4 f = p[i];
        xv[4 * i + 0] = f.x; xv[4 * i + 1] = f.y;
        xv[4 * i + 2] = f.z; xv[4 * i + 3] = f.w;
        mx = fmaxf(mx, fabsf(f.x)); mx = fmaxf(mx, fabsf(f.y));
        mx = fmaxf(mx, fabsf(f.z)); mx = fmaxf(mx, fabsf(f.w));
    }

    __shared__ float red[256];
    red[t] = mx;
    __syncthreads();
#pragma unroll
    for (int s = 128; s > 0; s >>= 1) {
        if (t < s) red[t] = fmaxf(red[t], red[t + s]);
        __syncthreads();
    }
    const float rowmax = fmaxf(red[0], 1e-5f);
    const float s127 = 127.0f / rowmax;
    if (t == 0) inv_s[m] = rowmax * (1.0f / 127.0f);

    const int mb = m >> 4;
    const int rlane = m & 15;
#pragma unroll
    for (int run = 0; run < 2; ++run) {
        unsigned long long pack = 0ull;
#pragma unroll
        for (int e = 0; e < 8; ++e) {
            float q = rintf(xv[run * 8 + e] * s127);
            q = fminf(fmaxf(q, -128.0f), 127.0f);
            pack |= ((unsigned long long)(unsigned char)(signed char)(int)q) << (8 * e);
        }
        const int k    = t * 16 + run * 8;
        const int kc   = k >> 6;
        const int kk   = k & 63;
        const int g    = ((kk >> 5) << 1) | ((kk >> 3) & 1);
        const int half = (kk >> 4) & 1;
        const int lane = rlane + (half << 4);
        const size_t off = ((size_t)(mb * KC + kc) << 10) + (size_t)lane * 32 + g * 8;
        *(unsigned long long*)(qA + off) = pack;
    }
}

__global__ __launch_bounds__(256) void unpack_pack_B(const int* __restrict__ wpk,
                                                     unsigned char* __restrict__ wB) {
    const int gid  = blockIdx.x * 256 + threadIdx.x;
    const int tile = gid >> 5;
    const int lane = gid & 31;
    const int nb   = tile >> 6;
    const int kc   = tile & 63;
    const int col  = lane & 15;
    const int half = lane >> 4;
    const int n    = nb * 16 + col;
    const int* rowp = wpk + (size_t)n * (K / 4);

#pragma unroll
    for (int run = 0; run < 2; ++run) {
        const int kbase = kc * 64 + run * 32 + half * 16;
        unsigned int dw[4];
#pragma unroll
        for (int pk = 0; pk < 4; ++pk) {
            const unsigned int b = (unsigned int)rowp[(kbase >> 2) + pk];
            unsigned int d = 0;
#pragma unroll
            for (int i = 0; i < 4; ++i) {
                const int code = (int)((b >> (2 * i)) & 3u) - 1;
                d |= ((unsigned int)(unsigned char)(signed char)code) << (8 * i);
            }
            dw[pk] = d;
        }
        const size_t off = ((size_t)(nb * KC + kc) << 10) + (size_t)lane * 32 + run * 16;
        uint4 v; v.x = dw[0]; v.y = dw[1]; v.z = dw[2]; v.w = dw[3];
        *(uint4*)(wB + off) = v;
    }
}

__global__ __launch_bounds__(256) void gemm_iu8(const unsigned char* __restrict__ qA,
                                                const unsigned char* __restrict__ wB,
                                                const float* __restrict__ inv_s,
                                                const float* __restrict__ wscale,
                                                unsigned short* __restrict__ out) {
    const int lane = threadIdx.x & 31;
    const int wave = threadIdx.x >> 5;
    const int m0 = blockIdx.y * 128 + (wave >> 2) * 64;
    const int n0 = blockIdx.x * 256 + (wave & 3) * 64;

    v8i acc[4][4];
#pragma unroll
    for (int i = 0; i < 4; ++i)
#pragma unroll
        for (int j = 0; j < 4; ++j)
            acc[i][j] = (v8i){0, 0, 0, 0, 0, 0, 0, 0};

    const unsigned char* aBase = qA + (((size_t)(m0 >> 4) * KC) << 10) + (size_t)lane * 32;
    const unsigned char* bBase = wB + (((size_t)(n0 >> 4) * KC) << 10) + (size_t)lane * 32;

    for (int kc = 0; kc < KC; ++kc) {
        v8i a[4], b[4];
#pragma unroll
        for (int i = 0; i < 4; ++i)
            a[i] = *(const v8i*)(aBase + ((size_t)i * KC << 10) + ((size_t)kc << 10));
#pragma unroll
        for (int j = 0; j < 4; ++j)
            b[j] = *(const v8i*)(bBase + ((size_t)j * KC << 10) + ((size_t)kc << 10));
#pragma unroll
        for (int i = 0; i < 4; ++i)
#pragma unroll
            for (int j = 0; j < 4; ++j)
                acc[i][j] = __builtin_amdgcn_wmma_i32_16x16x64_iu8(
                     true, a[i],  true, b[j], acc[i][j],
                     false,  false);
    }

    const int half = lane >> 4;
    const int col  = lane & 15;
    const float wsl = wscale[col & 3];

    float sA[4][8];
#pragma unroll
    for (int i = 0; i < 4; ++i) {
        const float* sp = inv_s + m0 + i * 16 + half * 8;
#pragma unroll
        for (int r = 0; r < 8; ++r) sA[i][r] = sp[r];
    }

#pragma unroll
    for (int i = 0; i < 4; ++i) {
#pragma unroll
        for (int r = 0; r < 8; ++r) {
            const size_t rowoff = (size_t)(m0 + i * 16 + half * 8 + r) * N;
#pragma unroll
            for (int j = 0; j < 4; ++j) {
                const float v = (float)acc[i][j][r] * sA[i][r] * wsl;
                __builtin_nontemporal_store(f2bf(v), out + rowoff + n0 + j * 16 + col);
            }
        }
    }
}

extern "C" void kernel_launch(void* const* d_in, const int* in_sizes, int n_in,
                              void* d_out, int out_size, void* d_ws, size_t ws_size,
                              hipStream_t stream) {
    const float* x    = (const float*)d_in[0];
    const int*   wpk  = (const int*)d_in[1];
    const float* wsc  = (const float*)d_in[2];
    unsigned short* out = (unsigned short*)d_out;

    unsigned char* qA = (unsigned char*)d_ws;
    unsigned char* wB = qA + (size_t)M * K;
    float* inv_s      = (float*)(wB + (size_t)N * K);

    quant_pack_A<<<M, 256, 0, stream>>>(x, qA, inv_s);
    unpack_pack_B<<<(NTILES_B * 32) / 256, 256, 0, stream>>>(wpk, wB);
    gemm_iu8<<<dim3(N / 256, M / 128), 256, 0, stream>>>(qA, wB, inv_s, wsc, out);
}
